// CorrectedLinear_70317204570758
// MI455X (gfx1250) — matrix operand order repaired and checked by reading, not compiled, not run on hardware
//
#include <hip/hip_runtime.h>

// ---------------------------------------------------------------------------
// CorrectedLinear on MI455X (gfx1250, wave32)
//   out[m][o] = sum_k x[m][k] * (W[o][k] + sum_r V_r[k][r]*C[o][r]) + b[o]
// Strategy: fold rank-32 correction into W_eff (bf16, 2MB in d_ws) once, then
// a single bf16-WMMA GEMM with f32 accumulation, double-buffered LDS pipeline.
// ---------------------------------------------------------------------------

typedef __attribute__((ext_vector_type(16))) __bf16        v16bf;
typedef __attribute__((ext_vector_type(8)))  __bf16        v8bf;
typedef __attribute__((ext_vector_type(8)))  float         v8f;
typedef __attribute__((ext_vector_type(4)))  float         v4f;
typedef __attribute__((ext_vector_type(4)))  unsigned int  v4u;

#define D_IN    1024
#define D_OUT   1024
#define RANK    32
#define M_TOTAL 65536   // 8 * 8192

#define BM 128
#define BN 128
#define BK 32
#define NK (D_IN / BK)  // 32 K-steps
#define LDA 40   // padded LDS row stride (elements): 80B, keeps 16B alignment
#define LDB 40

// ---------------------------------------------------------------------------
// Kernel 1: W_eff[o][i] = bf16( W[o][i] + sum_r V_r[i][r] * C[o][r] )
// ~67 MFLOP total: negligible. Hardware bf16 convert via __bf16 cast.
// ---------------------------------------------------------------------------
__global__ __launch_bounds__(256) void build_weff_bf16(
    const float* __restrict__ W,
    const float* __restrict__ Vr,   // (d_in, r) row-major
    const float* __restrict__ C,    // (d_out, r) row-major
    __bf16* __restrict__ Weff)
{
    int idx = blockIdx.x * 256 + threadIdx.x;   // o*1024 + i
    int o = idx >> 10;
    int i = idx & 1023;
    float acc = W[idx];
    const float* vr = Vr + (size_t)i * RANK;
    const float* cr = C  + (size_t)o * RANK;
#pragma unroll
    for (int r = 0; r < RANK; ++r) acc += vr[r] * cr[r];
    Weff[idx] = (__bf16)acc;
}

// ---------------------------------------------------------------------------
// Kernel 2: out = x @ W_eff^T + b    (bf16 WMMA, f32 accumulate)
// 256 threads = 8 waves; block tile 128x128, BK=32.
// Wave grid 4(M) x 2(N): each wave -> 32x64 sub-tile = 2x4 WMMA accumulators.
// Double-buffered LDS: one barrier per K-step; stage(k+1) overlaps wmma(k).
// ---------------------------------------------------------------------------
__global__ __launch_bounds__(256) void gemm_corrected_bf16(
    const float*  __restrict__ X,      // (M_TOTAL, D_IN) f32
    const __bf16* __restrict__ Weff,   // (D_OUT, D_IN) bf16
    const float*  __restrict__ bias,   // (D_OUT)
    float*        __restrict__ Out)    // (M_TOTAL, D_OUT)
{
    __shared__ __bf16 LdsA[2][BM * LDA];  // x tiles (2 x 10 KB)
    __shared__ __bf16 LdsB[2][BN * LDB];  // W_eff tiles (2 x 10 KB)

    const int tid   = threadIdx.x;
    const int lane  = tid & 31;
    const int wave  = tid >> 5;
    const int lr    = lane & 15;   // 0..15
    const int lh    = lane >> 4;   // 0..1 (K-half select, contiguous k order used here)
    const int waveM = wave >> 1;   // 0..3 -> M offset *32
    const int waveN = wave & 1;    // 0..1 -> N offset *64

    const int blockN = blockIdx.x * BN;
    const int blockM = blockIdx.y * BM;

    // staging assignment: thread -> (row 0..127, 16-element K half)
    const int sRow  = tid >> 1;
    const int sHalf = (tid & 1) * 16;

    const float*  gaBase = X    + (size_t)(blockM + sRow) * D_IN + sHalf;
    const __bf16* gbBase = Weff + (size_t)(blockN + sRow) * D_IN + sHalf;

    v8f acc[2][4];
#pragma unroll
    for (int mi = 0; mi < 2; ++mi)
#pragma unroll
        for (int ni = 0; ni < 4; ++ni)
            acc[mi][ni] = (v8f)(0.0f);

    union V16 { v16bf v; v8bf h[2]; };

    // staging registers
    v4f a0, a1, a2, a3;
    v4u b0, b1;

    // ---- prologue: load & stage tile 0 ----
    {
        const float* ga = gaBase;
        a0 = *(const v4f*)(ga + 0);
        a1 = *(const v4f*)(ga + 4);
        a2 = *(const v4f*)(ga + 8);
        a3 = *(const v4f*)(ga + 12);
        const v4u* gb = (const v4u*)gbBase;
        b0 = gb[0];
        b1 = gb[1];

        v8bf p0, p1;
#pragma unroll
        for (int e = 0; e < 4; ++e) {
            p0[e]     = (__bf16)a0[e];
            p0[4 + e] = (__bf16)a1[e];
            p1[e]     = (__bf16)a2[e];
            p1[4 + e] = (__bf16)a3[e];
        }
        *(v8bf*)&LdsA[0][sRow * LDA + sHalf]     = p0;
        *(v8bf*)&LdsA[0][sRow * LDA + sHalf + 8] = p1;
        *(v4u*)&LdsB[0][sRow * LDB + sHalf]      = b0;
        *(v4u*)&LdsB[0][sRow * LDB + sHalf + 8]  = b1;
    }
    __syncthreads();

    for (int kb = 0; kb < NK; ++kb) {
        const int cur = kb & 1;
        const int nxt = cur ^ 1;
        const bool hasNext = (kb + 1) < NK;

        // ---- issue global loads for tile kb+1 (latency hidden by wmma) ----
        if (hasNext) {
            const int k1 = (kb + 1) * BK;
            const float* ga = gaBase + k1;
            a0 = *(const v4f*)(ga + 0);
            a1 = *(const v4f*)(ga + 4);
            a2 = *(const v4f*)(ga + 8);
            a3 = *(const v4f*)(ga + 12);
            const v4u* gb = (const v4u*)(gbBase + k1);
            b0 = gb[0];
            b1 = gb[1];
            if (kb + 2 < NK) {   // WGP-scope global_prefetch of the tile after
                __builtin_prefetch((const void*)(ga + BK), 0, 3);
                __builtin_prefetch((const void*)(gbBase + k1 + BK), 0, 3);
            }
        }

        // ---- fragment loads from LDS[cur] (ISA 16-bit A 16x32 / B 32x16) ----
        const __bf16* bufA = LdsA[cur];
        const __bf16* bufB = LdsB[cur];
        V16 afrag[2], bfrag[4];
#pragma unroll
        for (int mi = 0; mi < 2; ++mi) {
            const int row = waveM * 32 + mi * 16 + lr;
            // lane 0-15: K=0..15 ; lane 16-31: K=16..31 (contiguous per row)
            afrag[mi].h[0] = *(const v8bf*)&bufA[row * LDA + lh * 16];
            afrag[mi].h[1] = *(const v8bf*)&bufA[row * LDA + lh * 16 + 8];
        }
#pragma unroll
        for (int ni = 0; ni < 4; ++ni) {
            const int col = waveN * 64 + ni * 16 + lr;   // N = lane%16
            // lane 0-15: K=0..15 ; lane 16-31: K=16..31 (contiguous per column)
            bfrag[ni].h[0] = *(const v8bf*)&bufB[col * LDB + lh * 16];
            bfrag[ni].h[1] = *(const v8bf*)&bufB[col * LDB + lh * 16 + 8];
        }

        // ---- 8x v_wmma_f32_16x16x32_bf16 ----
#pragma unroll
        for (int mi = 0; mi < 2; ++mi)
#pragma unroll
            for (int ni = 0; ni < 4; ++ni)
                acc[mi][ni] = __builtin_amdgcn_wmma_f32_16x16x32_bf16(
                    false, afrag[mi].v, false, bfrag[ni].v,
                    (short)0, acc[mi][ni], false, false);

        // ---- convert + stage tile kb+1 into the other buffer ----
        if (hasNext) {
            v8bf p0, p1;
#pragma unroll
            for (int e = 0; e < 4; ++e) {
                p0[e]     = (__bf16)a0[e];
                p0[4 + e] = (__bf16)a1[e];
                p1[e]     = (__bf16)a2[e];
                p1[4 + e] = (__bf16)a3[e];
            }
            *(v8bf*)&LdsA[nxt][sRow * LDA + sHalf]     = p0;
            *(v8bf*)&LdsA[nxt][sRow * LDA + sHalf + 8] = p1;
            *(v4u*)&LdsB[nxt][sRow * LDB + sHalf]      = b0;
            *(v4u*)&LdsB[nxt][sRow * LDB + sHalf + 8]  = b1;
            __syncthreads();   // nxt visible; cur free for next iteration's store
        }
    }

    // ---- epilogue: + bias, store f32 ----
    float bb[4];
#pragma unroll
    for (int ni = 0; ni < 4; ++ni)
        bb[ni] = bias[blockN + waveN * 64 + ni * 16 + lr];

#pragma unroll
    for (int mi = 0; mi < 2; ++mi) {
        const int rbase = blockM + waveM * 32 + mi * 16 + lh * 8; // VGPR v -> row rbase+v
#pragma unroll
        for (int ni = 0; ni < 4; ++ni) {
            const int col = blockN + waveN * 64 + ni * 16 + lr;
            float* op = Out + (size_t)rbase * D_OUT + col;
#pragma unroll
            for (int v = 0; v < 8; ++v)
                op[(size_t)v * D_OUT] = acc[mi][ni][v] + bb[ni];
        }
    }
}

// ---------------------------------------------------------------------------
extern "C" void kernel_launch(void* const* d_in, const int* in_sizes, int n_in,
                              void* d_out, int out_size, void* d_ws, size_t ws_size,
                              hipStream_t stream) {
    (void)in_sizes; (void)n_in; (void)out_size; (void)ws_size;
    const float* x  = (const float*)d_in[0];   // (8, 8192, 1024)
    const float* W  = (const float*)d_in[1];   // (1024, 1024)
    const float* b  = (const float*)d_in[2];   // (1024)
    const float* Vr = (const float*)d_in[3];   // (1024, 32)
    const float* C  = (const float*)d_in[4];   // (1024, 32)
    float* out = (float*)d_out;
    __bf16* Weff = (__bf16*)d_ws;              // 1024*1024 bf16 = 2 MB scratch

    build_weff_bf16<<<(D_OUT * D_IN) / 256, 256, 0, stream>>>(W, Vr, C, Weff);

    dim3 grid(D_OUT / BN, M_TOTAL / BM);   // N-fastest -> L2 reuse of x stripes
    gemm_corrected_bf16<<<grid, 256, 0, stream>>>(x, Weff, b, out);
}
